// upsample_17755394802118
// MI455X (gfx1250) — matrix operand order repaired and checked by reading, not compiled, not run on hardware
//
#include <hip/hip_runtime.h>
#include <hip/hip_bf16.h>

typedef __attribute__((ext_vector_type(16))) __bf16 v16bf;
typedef __attribute__((ext_vector_type(8)))  float  v8f;

#define C_IN   128
#define C_OUT  64
#define HW     128
#define HWO    256
#define PK     72          // LDS k-stride: 144B row; fragment chunks 16B aligned
#define XROWS  130         // x = -1 .. 128 (guard rows 0 and 129 stay zero)
#define PER_IMG 4194304    // 64*256*256

__device__ __forceinline__ unsigned short f2bf(float f) {
    unsigned int u = __float_as_uint(f);
    unsigned int r = u + 0x7FFFu + ((u >> 16) & 1u);   // round-to-nearest-even
    return (unsigned short)(r >> 16);
}

__device__ __forceinline__ unsigned pk_bf16(float a, float b) {
#if __has_builtin(__builtin_amdgcn_cvt_pk_bf16_f32)
    auto h = __builtin_amdgcn_cvt_pk_bf16_f32(a, b);   // one v_cvt_pk_bf16_f32
    return __builtin_bit_cast(unsigned, h);
#else
    __hip_bfloat162 h = __float22bfloat162_rn(make_float2(a, b));
    return *reinterpret_cast<unsigned*>(&h);
#endif
}

// ---------------------------------------------------------------------------
// Kernel 0: pack weights [128,64,4,4] fp32 into bf16 WMMA A-fragments.
// Fragment order: [cls(4)][kstep(16)][mtile(4)][lane(32)][dword(8)]  (256 KB)
// A packing used here (contiguous k, B matches): lanes 0-15 = rows M, K=kk+{0..15};
//                      lanes 16-31 = rows M, K=kk+{16..31}.
// K index = tap*128 + ic, tap = dy*2+dx; kh = 1-r+2*dy, kw = 1-c+2*dx.
// ---------------------------------------------------------------------------
__global__ void pack_w(const float* __restrict__ w, unsigned int* __restrict__ afrag) {
    int d = blockIdx.x * blockDim.x + threadIdx.x;     // 0..65535 dwords
    int dw     = d & 7;
    int lane   = (d >> 3) & 31;
    int mt     = (d >> 8) & 3;
    int kstepg = (d >> 10) & 15;
    int cls    = (d >> 14) & 3;
    int kk = kstepg * 32;
    int oc = mt * 16 + (lane & 15);
    int r = cls >> 1, c = cls & 1;
    unsigned short e[2];
#pragma unroll
    for (int t = 0; t < 2; ++t) {
        int i = 2 * dw + t;
        int K = kk + ((lane < 16) ? 0 : 16) + i;
        int tap = K >> 7, ic = K & 127;
        int dy = tap >> 1, dx = tap & 1;
        int kh = 1 - r + 2 * dy;
        int kw = 1 - c + 2 * dx;
        e[t] = f2bf(w[((ic * C_OUT + oc) * 4 + kh) * 4 + kw]);
    }
    afrag[d] = (unsigned int)e[0] | ((unsigned int)e[1] << 16);
}

// ---------------------------------------------------------------------------
// Kernel 1: implicit-GEMM deconv. grid = (y:128, img:16, cls:4), block = 256.
// 4 staging chunks (dy x ic-half); each chunk's tile serves BOTH dx taps via
// a +-1 row shift of the B-fragment base (guard rows at x=-1/128 are zero).
// Double-buffered: chunk ch+1 loads/stores overlap chunk ch's 16 WMMAs.
// ---------------------------------------------------------------------------
__global__ void __launch_bounds__(256)
deconv_gemm(const float* __restrict__ xin,
            const unsigned int* __restrict__ afrag,
            float* __restrict__ out) {
    __shared__ __align__(32) unsigned int   aw[16384];          // 64 KB A frags
    __shared__ __align__(32) unsigned short pt[2][XROWS * PK];  // 2 x 18.7 KB B tiles

    const int y   = blockIdx.x;
    const int n   = blockIdx.y;
    const int cls = blockIdx.z;
    const int r = cls >> 1, c = cls & 1;
    const int tid  = threadIdx.x;
    const int lane = tid & 31;
    const int wave = tid >> 5;
    const int wm = wave >> 2;                 // 0..1: mtile pair {2wm, 2wm+1}
    const int wn = wave & 3;                  // 0..3: col block of 32
    const int colbase = wn * 32 + (lane & 15);
    const int khalf = (lane >> 4) << 4;       // 0 or 16

    // ---- async DMA: 64 KB of A fragments global -> LDS (16B/lane/issue) ----
    {
        const char* gsrc = (const char*)(afrag + cls * 16384) + tid * 16;
        unsigned    ldst = (unsigned)(unsigned long long)(void*)((char*)aw + tid * 16);
#pragma unroll
        for (int i = 0; i < 16; ++i) {
            asm volatile("global_load_async_to_lds_b128 %0, %1, off"
                         :: "v"(ldst + (unsigned)(i * 4096)),
                            "v"(gsrc + i * 4096)
                         : "memory");
        }
        asm volatile("s_wait_asynccnt 0" ::: "memory");
    }

    // zero guard rows (x=-1 and x=128) of both buffers; never written again
    if (tid < 64) {
        int b = tid >> 5, rw = (tid >> 4) & 1, seg = tid & 15;
        unsigned long long* g = (unsigned long long*)
            &pt[b][(rw ? (XROWS - 1) : 0) * PK + seg * 4];
        *g = 0ull;
    }

    float4 rq[8];   // staged fp32: 8 quads (4 ic x 1 x) per thread per chunk

    // chunk ch = (dy<<1)|ichalf: rows ih(dy), ic [ichalf*64, +64), all 128 x
    auto ldchunk = [&](int ch) {
        const int dy = ch >> 1, icb = (ch & 1) << 6;
        const int ih = y + ((r == 0) ? -dy : (1 - dy));
        const int ihc = min(max(ih, 0), HW - 1);
        const float* src = xin + (((size_t)(n * C_IN + icb)) * HW + ihc) * HW;
#pragma unroll
        for (int i = 0; i < 8; ++i) {
            int q   = tid + (i << 8);             // 0..2047 quads
            int icl = ((q >> 7) << 2);            // 0,4,...,60
            int xx  = q & 127;
            const float* p = src + (size_t)icl * (HW * HW) + xx;
            rq[i].x = p[0 * HW * HW];
            rq[i].y = p[1 * HW * HW];
            rq[i].z = p[2 * HW * HW];
            rq[i].w = p[3 * HW * HW];
        }
    };
    auto stchunk = [&](unsigned short* buf, int ch) {
        const int dy = ch >> 1;
        const int ih = y + ((r == 0) ? -dy : (1 - dy));
        const bool rowok = (unsigned)ih < (unsigned)HW;
#pragma unroll
        for (int i = 0; i < 8; ++i) {
            int q   = tid + (i << 8);
            int icl = ((q >> 7) << 2);
            int xx  = q & 127;
            unsigned lo = pk_bf16(fmaxf(rq[i].x, 0.f), fmaxf(rq[i].y, 0.f));
            unsigned hi = pk_bf16(fmaxf(rq[i].z, 0.f), fmaxf(rq[i].w, 0.f));
            lo = rowok ? lo : 0u;                 // v_cndmask, no branch
            hi = rowok ? hi : 0u;
            unsigned* dst = (unsigned*)&buf[(xx + 1) * PK + icl];   // 8B aligned
            dst[0] = lo;
            dst[1] = hi;
        }
    };

    v8f acc[4] = {};   // [mi*2+nc]

    auto compute = [&](const unsigned short* buf, int ch) {
        const int dy = ch >> 1, ichalf = ch & 1;
#pragma unroll
        for (int dx = 0; dx < 2; ++dx) {
            const int tap = dy * 2 + dx;
            const int ax  = (c == 0) ? -dx : (1 - dx);
            v16bf af[2][2], bf[2][2];
#pragma unroll
            for (int ks = 0; ks < 2; ++ks) {
                const int kstepg = tap * 4 + ichalf * 2 + ks;
#pragma unroll
                for (int mi = 0; mi < 2; ++mi)
                    af[ks][mi] = *(const v16bf*)
                        &aw[((kstepg * 4 + wm * 2 + mi) * 32 + lane) * 8];
#pragma unroll
                for (int nc = 0; nc < 2; ++nc)
                    bf[ks][nc] = *(const v16bf*)
                        &buf[(colbase + nc * 16 + ax + 1) * PK + ks * 32 + khalf];
            }
#pragma unroll
            for (int ks = 0; ks < 2; ++ks)
#pragma unroll
                for (int mi = 0; mi < 2; ++mi)
#pragma unroll
                    for (int nc = 0; nc < 2; ++nc)
                        acc[mi * 2 + nc] = __builtin_amdgcn_wmma_f32_16x16x32_bf16(
                            false, af[ks][mi], false, bf[ks][nc],
                            (short)0, acc[mi * 2 + nc], false, false);
        }
    };

    ldchunk(0);
    stchunk(pt[0], 0);
    for (int ch = 0; ch < 4; ++ch) {
        __syncthreads();                  // buf[ch&1] + guards visible to all
        if (ch < 3) ldchunk(ch + 1);      // long-latency global loads first
        compute(pt[ch & 1], ch);          // 16 WMMAs (both dx taps)
        if (ch < 3) stchunk(pt[(ch + 1) & 1], ch + 1);  // other buffer: no race
    }

    // D 16x16 f32 layout: VGPR j, lanes 0-15: M=j, N=lane; lanes 16-31: M=j+8.
    const int oh = 2 * y + r;
    const int ocb = (lane >> 4) << 3;             // 0 or 8
#pragma unroll
    for (int mi = 0; mi < 2; ++mi)
#pragma unroll
        for (int nc = 0; nc < 2; ++nc) {
            const int ow = 2 * (colbase + nc * 16) + c;
            float* ob = out + (((size_t)(n * C_OUT)) * HWO + oh) * HWO + ow;
#pragma unroll
            for (int j = 0; j < 8; ++j) {
                int oc = (wm * 2 + mi) * 16 + ocb + j;
                ob[(size_t)oc * (HWO * HWO)] = acc[mi * 2 + nc][j];
            }
        }
}

// ---------------------------------------------------------------------------
// GroupNorm: deterministic two-stage reduction (no float atomics).
// ---------------------------------------------------------------------------
__global__ void reduce1(const float* __restrict__ out, float* __restrict__ part) {
    const int n = blockIdx.y, b = blockIdx.x;     // (1024, 16)
    const float4* p = (const float4*)(out + (size_t)n * PER_IMG + (size_t)b * 4096);
    float s = 0.f, s2 = 0.f;
#pragma unroll
    for (int k = 0; k < 4; ++k) {
        float4 v = p[threadIdx.x + (k << 8)];
        s  += v.x + v.y + v.z + v.w;
        s2 += v.x * v.x + v.y * v.y + v.z * v.z + v.w * v.w;
    }
    __shared__ float sh0[256], sh1[256];
    sh0[threadIdx.x] = s; sh1[threadIdx.x] = s2;
    __syncthreads();
    for (int off = 128; off > 0; off >>= 1) {
        if (threadIdx.x < off) {
            sh0[threadIdx.x] += sh0[threadIdx.x + off];
            sh1[threadIdx.x] += sh1[threadIdx.x + off];
        }
        __syncthreads();
    }
    if (threadIdx.x == 0) {
        part[n * 1024 + b]             = sh0[0];
        part[16 * 1024 + n * 1024 + b] = sh1[0];
    }
}

__global__ void reduce2(const float* __restrict__ part, float* __restrict__ stats) {
    const int n = blockIdx.x;                     // 16 blocks
    float s = 0.f, s2 = 0.f;
    for (int i = threadIdx.x; i < 1024; i += 256) {
        s  += part[n * 1024 + i];
        s2 += part[16384 + n * 1024 + i];
    }
    __shared__ float sh0[256], sh1[256];
    sh0[threadIdx.x] = s; sh1[threadIdx.x] = s2;
    __syncthreads();
    for (int off = 128; off > 0; off >>= 1) {
        if (threadIdx.x < off) {
            sh0[threadIdx.x] += sh0[threadIdx.x + off];
            sh1[threadIdx.x] += sh1[threadIdx.x + off];
        }
        __syncthreads();
    }
    if (threadIdx.x == 0) {
        float mean = sh0[0] * (1.f / (float)PER_IMG);
        float var  = sh1[0] * (1.f / (float)PER_IMG) - mean * mean;
        stats[n * 2]     = mean;
        stats[n * 2 + 1] = rsqrtf(var + 1e-5f);
    }
}

__global__ void norm_k(float* __restrict__ out, const float* __restrict__ stats) {
    const int n = blockIdx.x >> 10;               // 1024 blocks per image
    const float mean = stats[n * 2];
    const float inv  = stats[n * 2 + 1];
    float4* p = (float4*)(out + (size_t)blockIdx.x * 4096);
#pragma unroll
    for (int k = 0; k < 4; ++k) {
        int i = threadIdx.x + (k << 8);
        float4 v = p[i];
        v.x = (v.x - mean) * inv;
        v.y = (v.y - mean) * inv;
        v.z = (v.z - mean) * inv;
        v.w = (v.w - mean) * inv;
        p[i] = v;
    }
}

// ---------------------------------------------------------------------------
extern "C" void kernel_launch(void* const* d_in, const int* in_sizes, int n_in,
                              void* d_out, int out_size, void* d_ws, size_t ws_size,
                              hipStream_t stream) {
    const float* x = (const float*)d_in[0];        // (16,128,128,128) fp32
    const float* w = (const float*)d_in[1];        // (128,64,4,4) fp32
    float* out = (float*)d_out;                    // (16,64,256,256) fp32

    // ws layout: [0,256KB) A-fragments | [256KB,+128KB) partials | then stats
    unsigned int* afrag = (unsigned int*)d_ws;
    float* part  = (float*)((char*)d_ws + 65536u * 4u);
    float* stats = (float*)((char*)d_ws + 65536u * 4u + 32768u * 4u);

    pack_w<<<dim3(256), dim3(256), 0, stream>>>(w, afrag);
    deconv_gemm<<<dim3(128, 16, 4), dim3(256), 0, stream>>>(x, afrag, out);
    reduce1<<<dim3(1024, 16), dim3(256), 0, stream>>>(out, part);
    reduce2<<<dim3(16), dim3(256), 0, stream>>>(part, stats);
    norm_k<<<dim3(16384), dim3(256), 0, stream>>>(out, stats);
}
